// BasicVSR_15564961480894
// MI455X (gfx1250) — matrix operand order repaired and checked by reading, not compiled, not run on hardware
//
#include <hip/hip_runtime.h>
#include <math.h>

typedef __attribute__((ext_vector_type(16))) __bf16 bf16x16;
typedef __attribute__((ext_vector_type(8)))  __bf16 bf16x8;
typedef __attribute__((ext_vector_type(8)))  float  f32x8;
typedef __attribute__((ext_vector_type(4)))  unsigned u32x4;

#define HWN 4096

static __device__ __forceinline__ float apply_act(float v, int act) {
  if (act == 1) return v >= 0.f ? v : 0.1f * v;          // lrelu 0.1
  if (act == 2) return v > 0.f ? v : 0.f;                // relu
  if (act == 3) return 0.5f * v * (1.f + erff(v * 0.70710678118654752f)); // exact gelu
  return v;
}

static __device__ __forceinline__ bf16x16 cat8(bf16x8 a, bf16x8 b) {
  return __builtin_shufflevector(a, b, 0,1,2,3,4,5,6,7,8,9,10,11,12,13,14,15);
}

// generic -> LDS(as3) offset for async-to-LDS destination VGPR
typedef __attribute__((address_space(3))) unsigned char lds_byte;
static __device__ __forceinline__ unsigned lds_addr_of(const void* p) {
  return (unsigned)(unsigned long long)(const lds_byte*)p;
}

// ---------------------------------------------------------------------------
// 3x3 SAME conv as implicit GEMM on v_wmma_f32_16x16x32_bf16.
// Two N-tiles (32 px) per wave share one A fragment -> 2 independent WMMA
// chains (hides XDL latency) and half the A-fragment traffic.
// in:   [H][W][Cinp] bf16 (Cinp mult of 32, pad channels zeroed)
// wp:   [Coutp][9*Cinp] bf16, k = (ky*3+kx)*Cinp + ci
// outb: [H][W][ldo] bf16 (pads zeroed) | outf: [Cout][H*W] f32 planar
// res:  optional residual read from [H][W][ldo]
// Halo tile staged into LDS with GLOBAL_LOAD_ASYNC_TO_LDS_B128 (ASYNCcnt).
// ---------------------------------------------------------------------------
template <int ACT, bool RES, bool OB, bool OF>
__global__ __launch_bounds__(256) void conv3x3_wmma(
    const __bf16* __restrict__ in, const __bf16* __restrict__ wp,
    const float* __restrict__ bias, __bf16* __restrict__ outb,
    float* __restrict__ outf, const __bf16* __restrict__ res,
    int H, int W, int Cinp, int Cout, int Coutp, int ldo)
{
  extern __shared__ __bf16 smem[];               // [3][66][Cinp]
  const int SW = 64;
  const int nstrip = W >> 6;
  const int y = blockIdx.x / nstrip;
  const int xbase = (blockIdx.x - y * nstrip) << 6;
  const int K = 9 * Cinp;
  const int rowPitch = (SW + 2) * Cinp;

  // --- async halo staging: interior cells via async-to-LDS, OOB cells zero ---
  {
    const int chunks = Cinp >> 3;                // 16B (8 x bf16) chunks per pixel
    const int total = 3 * (SW + 2) * chunks;
    const unsigned lbase = lds_addr_of(smem);
    for (int t = threadIdx.x; t < total; t += 256) {
      int c8 = (t % chunks) << 3;
      int rest = t / chunks;
      int xx = rest % (SW + 2);
      int ry = rest / (SW + 2);
      int gy = y + ry - 1;
      int gx = xbase + xx - 1;
      unsigned loff = (unsigned)((ry * rowPitch + xx * Cinp + c8) * 2);
      if (gy >= 0 && gy < H && gx >= 0 && gx < W) {
        unsigned laddr = lbase + loff;
        unsigned long long ga =
            (unsigned long long)(const void*)(in + ((size_t)gy * W + gx) * Cinp + c8);
        asm volatile("global_load_async_to_lds_b128 %0, %1, off"
                     :: "v"(laddr), "v"(ga) : "memory");
      } else {
        *(u32x4*)((char*)smem + loff) = (u32x4){0u, 0u, 0u, 0u};
      }
    }
    asm volatile("s_wait_asynccnt 0x0" ::: "memory");
  }
  __syncthreads();

  const int wave = threadIdx.x >> 5;
  const int lane = threadIdx.x & 31;
  const int half = lane >> 4;
  const int l16 = lane & 15;
  const int Mt = Coutp >> 4;
  const int nTiles = Mt * 2;                      // pairs of N-tiles per strip

  for (int tile = wave; tile < nTiles; tile += 8) {
    const int mt = tile >> 1;
    const int np = tile & 1;
    const int mbase = mt * 16 + half * 8;         // C/D rows owned by this lane
    float bsum[8];
#pragma unroll
    for (int r = 0; r < 8; ++r)
      bsum[r] = (mbase + r < Cout) ? bias[mbase + r] : 0.f;

    f32x8 acc0 = {}, acc1 = {};
    const __bf16* wrow = wp + (size_t)(mt * 16 + l16) * K + half * 8;
    const int xl0 = np * 32 + l16;                // first N-tile pixel column
#pragma unroll
    for (int kidx = 0; kidx < 9; ++kidx) {
      const int ky = kidx / 3;                    // constants after unroll
      const int kx = kidx - ky * 3;
      const __bf16* wk = wrow + kidx * Cinp;
      const __bf16* brow0 = smem + (size_t)(ky * (SW + 2) + xl0 + kx) * Cinp + half * 8;
      const __bf16* brow1 = brow0 + 16 * Cinp;    // second N-tile, +16 pixels
      for (int c0 = 0; c0 < Cinp; c0 += 32) {
        bf16x8 a0 = *(const bf16x8*)(wk + c0);
        bf16x8 a1 = *(const bf16x8*)(wk + c0 + 16);
        bf16x16 A = cat8(a0, a1);
        bf16x8 b00 = *(const bf16x8*)(brow0 + c0);
        bf16x8 b01 = *(const bf16x8*)(brow0 + c0 + 16);
        bf16x8 b10 = *(const bf16x8*)(brow1 + c0);
        bf16x8 b11 = *(const bf16x8*)(brow1 + c0 + 16);
        acc0 = __builtin_amdgcn_wmma_f32_16x16x32_bf16(false, A, false, cat8(b00, b01),
                                                       (short)0, acc0, false, false);
        acc1 = __builtin_amdgcn_wmma_f32_16x16x32_bf16(false, A, false, cat8(b10, b11),
                                                       (short)0, acc1, false, false);
      }
    }

#pragma unroll
    for (int u = 0; u < 2; ++u) {
      const f32x8& acc = u ? acc1 : acc0;
      const int x = xbase + xl0 + u * 16;
      const size_t pix = (size_t)y * W + x;
      bf16x8 rv;
      if constexpr (RES) rv = *(const bf16x8*)(res + pix * ldo + mbase);
      float vr[8];
#pragma unroll
      for (int r = 0; r < 8; ++r) {
        float v = acc[r] + bsum[r];
        if constexpr (RES) v += (float)rv[r];
        if constexpr (ACT == 1) v = v >= 0.f ? v : 0.1f * v;
        if constexpr (ACT == 2) v = fmaxf(v, 0.f);
        vr[r] = (mbase + r < Cout) ? v : 0.f;     // keep channel padding zeroed
      }
      if constexpr (OB) {
        bf16x8 ov;
#pragma unroll
        for (int r = 0; r < 8; ++r) ov[r] = (__bf16)vr[r];
        *(bf16x8*)(outb + pix * ldo + mbase) = ov;  // one 16B global_store
      }
      if constexpr (OF) {
#pragma unroll
        for (int r = 0; r < 8; ++r)
          if (mbase + r < Cout) outf[(size_t)(mbase + r) * H * W + pix] = vr[r];
      }
    }
  }
}

// ---------------------------------------------------------------------------
// Flat GEMM: out[M,N] = Wt[Mp,Kp] * X[N,Kp]^T ; X token-major bf16.
// Two N-tiles per wave share the A fragment. outb token-major [N][ldo]+co_off.
// ---------------------------------------------------------------------------
template <int ACT>
__global__ __launch_bounds__(256) void gemm_wmma(
    const __bf16* __restrict__ X, const __bf16* __restrict__ Wt,
    const float* __restrict__ bias, __bf16* __restrict__ outb,
    int N, int Kp, int M, int Mp, int ldo, int co_off)
{
  const int gwave = (blockIdx.x * 256 + threadIdx.x) >> 5;
  const int lane = threadIdx.x & 31;
  const int Ntt = N >> 5;                          // N-tile pairs
  const int Mtt = Mp >> 4;
  if (gwave >= Ntt * Mtt) return;
  const int mt = gwave / Ntt;
  const int nt = gwave - mt * Ntt;
  const int half = lane >> 4;
  const int l16 = lane & 15;
  const int mbase = mt * 16 + half * 8;
  float bsum[8];
#pragma unroll
  for (int r = 0; r < 8; ++r)
    bsum[r] = (mbase + r < M) ? bias[mbase + r] : 0.f;

  const __bf16* wrow = Wt + (size_t)(mt * 16 + l16) * Kp + half * 8;
  const __bf16* xrow0 = X + (size_t)(nt * 32 + l16) * Kp + half * 8;
  const __bf16* xrow1 = xrow0 + (size_t)16 * Kp;
  f32x8 acc0 = {}, acc1 = {};
  for (int k0 = 0; k0 < Kp; k0 += 32) {
    bf16x8 a0 = *(const bf16x8*)(wrow + k0);
    bf16x8 a1 = *(const bf16x8*)(wrow + k0 + 16);
    bf16x16 A = cat8(a0, a1);
    bf16x8 b00 = *(const bf16x8*)(xrow0 + k0);
    bf16x8 b01 = *(const bf16x8*)(xrow0 + k0 + 16);
    bf16x8 b10 = *(const bf16x8*)(xrow1 + k0);
    bf16x8 b11 = *(const bf16x8*)(xrow1 + k0 + 16);
    acc0 = __builtin_amdgcn_wmma_f32_16x16x32_bf16(false, A, false, cat8(b00, b01),
                                                   (short)0, acc0, false, false);
    acc1 = __builtin_amdgcn_wmma_f32_16x16x32_bf16(false, A, false, cat8(b10, b11),
                                                   (short)0, acc1, false, false);
  }
#pragma unroll
  for (int u = 0; u < 2; ++u) {
    const f32x8& acc = u ? acc1 : acc0;
    const int n = nt * 32 + u * 16 + l16;
#pragma unroll
    for (int r = 0; r < 8; ++r) {
      const int m = mbase + r;
      float v = apply_act(acc[r] + bsum[r], ACT);
      if (m >= M) v = 0.f;
      outb[(size_t)n * ldo + co_off + m] = (__bf16)v;
    }
  }
}

// ------------------------------ packing ------------------------------------
__global__ void pack_conv(const float* __restrict__ w, __bf16* __restrict__ dst,
                          int Cout, int Cin, int Coutp, int Cinp) {
  int t = blockIdx.x * blockDim.x + threadIdx.x;
  int K = 9 * Cinp;
  if (t >= Coutp * K) return;
  int co = t / K;
  int rem = t - co * K;
  int kidx = rem / Cinp;
  int ci = rem - kidx * Cinp;
  float v = (co < Cout && ci < Cin) ? w[(size_t)(co * Cin + ci) * 9 + kidx] : 0.f;
  dst[t] = (__bf16)v;
}

__global__ void pack_dcn(const float* __restrict__ w, __bf16* __restrict__ dst) {
  // dcn_w [64][64][3][3] -> [64][576], k = (g*9+kk)*16 + c, src channel g*16+c
  int t = blockIdx.x * blockDim.x + threadIdx.x;
  if (t >= 64 * 576) return;
  int co = t / 576;
  int rem = t - co * 576;
  int j = rem >> 4;           // g*9+kk
  int c = rem & 15;
  int g = j / 9;
  int kk = j - g * 9;
  dst[t] = (__bf16)w[(size_t)(co * 64 + g * 16 + c) * 9 + kk];
}

__global__ void pack_mat(const float* __restrict__ src, __bf16* __restrict__ dst,
                         int M, int K, int Mp, int Kp, int rs, int cs) {
  int t = blockIdx.x * blockDim.x + threadIdx.x;
  if (t >= Mp * Kp) return;
  int m = t / Kp;
  int k = t - m * Kp;
  float v = (m < M && k < K) ? src[(size_t)m * rs + (size_t)k * cs] : 0.f;
  dst[t] = (__bf16)v;
}

// ---------------------------- elementwise ----------------------------------
__global__ void zero_bf16k(__bf16* __restrict__ p, int n) {
  int t = blockIdx.x * blockDim.x + threadIdx.x;
  if (t < n) p[t] = (__bf16)0.f;
}

__global__ void f2h_hwc(const float* __restrict__ src, __bf16* __restrict__ dst,
                        int C, int HWn, int ldo, int offc) {
  int t = blockIdx.x * blockDim.x + threadIdx.x;
  if (t >= C * HWn) return;
  int c = t / HWn;
  int p = t - c * HWn;
  dst[(size_t)p * ldo + offc + c] = (__bf16)src[t];
}

__global__ void hwc_copy(const __bf16* __restrict__ src, __bf16* __restrict__ dst,
                         int C, int HWn, int lds_, int soff, int ldd, int doff) {
  int t = blockIdx.x * blockDim.x + threadIdx.x;
  if (t >= C * HWn) return;
  int p = t / C;
  int c = t - p * C;
  dst[(size_t)p * ldd + doff + c] = src[(size_t)p * lds_ + soff + c];
}

__global__ void warp_bilinear(const __bf16* __restrict__ feat, const float* __restrict__ flow,
                              __bf16* __restrict__ outp, int H, int W, int C, int ldf, int ldo) {
  int t = blockIdx.x * blockDim.x + threadIdx.x;
  int groups = C >> 3;
  if (t >= H * W * groups) return;
  int p = t / groups;
  int cb = (t - p * groups) << 3;
  int py = p / W, px = p - py * W;
  float xs = (float)px + flow[p];            // ch0 = x-flow
  float ys = (float)py + flow[H * W + p];    // ch1 = y-flow
  float x0f = floorf(xs), y0f = floorf(ys);
  int x0 = (int)x0f, y0 = (int)y0f;
  float wx = xs - x0f, wy = ys - y0f;
  bool vy0 = (y0 >= 0) && (y0 < H);
  bool vy1 = (y0 + 1 >= 0) && (y0 + 1 < H);
  bool vx0 = (x0 >= 0) && (x0 < W);
  bool vx1 = (x0 + 1 >= 0) && (x0 + 1 < W);
  int y0c = y0 < 0 ? 0 : (y0 >= H ? H - 1 : y0);
  int y1c = y0 + 1 < 0 ? 0 : (y0 + 1 >= H ? H - 1 : y0 + 1);
  int x0c = x0 < 0 ? 0 : (x0 >= W ? W - 1 : x0);
  int x1c = x0 + 1 < 0 ? 0 : (x0 + 1 >= W ? W - 1 : x0 + 1);
  float w00 = (vy0 && vx0) ? (1.f - wy) * (1.f - wx) : 0.f;
  float w01 = (vy0 && vx1) ? (1.f - wy) * wx : 0.f;
  float w10 = (vy1 && vx0) ? wy * (1.f - wx) : 0.f;
  float w11 = (vy1 && vx1) ? wy * wx : 0.f;
  const __bf16* p00 = feat + ((size_t)y0c * W + x0c) * ldf + cb;
  const __bf16* p01 = feat + ((size_t)y0c * W + x1c) * ldf + cb;
  const __bf16* p10 = feat + ((size_t)y1c * W + x0c) * ldf + cb;
  const __bf16* p11 = feat + ((size_t)y1c * W + x1c) * ldf + cb;
  __bf16* dst = outp + (size_t)p * ldo + cb;
#pragma unroll
  for (int j = 0; j < 8; ++j) {
    float v = w00 * (float)p00[j] + w01 * (float)p01[j] +
              w10 * (float)p10[j] + w11 * (float)p11[j];
    dst[j] = (__bf16)v;
  }
}

__global__ void dcn_sample(const __bf16* __restrict__ xf, const float* __restrict__ o,
                           const float* __restrict__ flow, __bf16* __restrict__ samp,
                           int H, int W) {
  int t = blockIdx.x * blockDim.x + threadIdx.x;
  int HWn = H * W;
  if (t >= HWn * 36) return;
  int p = t / 36;
  int gk = t - p * 36;
  int g = gk / 9;
  int k = gk - g * 9;
  int py = p / W, px = p - py * W;
  // offset channel order: c = g*18 + k*2 + comp ; comp0 += y-flow (reversed tile)
  float offy = 10.f * tanhf(o[(size_t)(g * 18 + k * 2 + 0) * HWn + p]) + flow[HWn + p];
  float offx = 10.f * tanhf(o[(size_t)(g * 18 + k * 2 + 1) * HWn + p]) + flow[p];
  float msk = 1.f / (1.f + expf(-o[(size_t)(72 + g * 9 + k) * HWn + p]));
  float ys = (float)(py + k / 3 - 1) + offy;
  float xs = (float)(px + (k % 3) - 1) + offx;
  float y0f = floorf(ys), x0f = floorf(xs);
  int y0 = (int)y0f, x0 = (int)x0f;
  float wy = ys - y0f, wx = xs - x0f;
  bool vy0 = (y0 >= 0) && (y0 < H);
  bool vy1 = (y0 + 1 >= 0) && (y0 + 1 < H);
  bool vx0 = (x0 >= 0) && (x0 < W);
  bool vx1 = (x0 + 1 >= 0) && (x0 + 1 < W);
  int y0c = y0 < 0 ? 0 : (y0 >= H ? H - 1 : y0);
  int y1c = y0 + 1 < 0 ? 0 : (y0 + 1 >= H ? H - 1 : y0 + 1);
  int x0c = x0 < 0 ? 0 : (x0 >= W ? W - 1 : x0);
  int x1c = x0 + 1 < 0 ? 0 : (x0 + 1 >= W ? W - 1 : x0 + 1);
  float w00 = (vy0 && vx0) ? (1.f - wy) * (1.f - wx) : 0.f;
  float w01 = (vy0 && vx1) ? (1.f - wy) * wx : 0.f;
  float w10 = (vy1 && vx0) ? wy * (1.f - wx) : 0.f;
  float w11 = (vy1 && vx1) ? wy * wx : 0.f;
  int cb = g * 16;
  const __bf16* p00 = xf + ((size_t)y0c * W + x0c) * 64 + cb;
  const __bf16* p01 = xf + ((size_t)y0c * W + x1c) * 64 + cb;
  const __bf16* p10 = xf + ((size_t)y1c * W + x0c) * 64 + cb;
  const __bf16* p11 = xf + ((size_t)y1c * W + x1c) * 64 + cb;
  __bf16* dst = samp + (size_t)p * 576 + gk * 16;
#pragma unroll
  for (int j = 0; j < 16; ++j) {
    float v = w00 * (float)p00[j] + w01 * (float)p01[j] +
              w10 * (float)p10[j] + w11 * (float)p11[j];
    dst[j] = (__bf16)(msk * v);
  }
}

__global__ void moe_gate(const __bf16* __restrict__ tok, const float* __restrict__ gw,
                         float* __restrict__ comb, int N, int D, int ldt) {
  int t = blockIdx.x * blockDim.x + threadIdx.x;
  if (t >= N) return;
  const __bf16* tp = tok + (size_t)t * ldt;
  float l0 = 0.f, l1 = 0.f, l2 = 0.f, l3 = 0.f;
  for (int d = 0; d < D; ++d) {
    float x = (float)tp[d];
    l0 += x * gw[d * 4 + 0];
    l1 += x * gw[d * 4 + 1];
    l2 += x * gw[d * 4 + 2];
    l3 += x * gw[d * 4 + 3];
  }
  float mx = fmaxf(fmaxf(l0, l1), fmaxf(l2, l3));
  float e[4] = {expf(l0 - mx), expf(l1 - mx), expf(l2 - mx), expf(l3 - mx)};
  int i0 = 0;
  for (int k = 1; k < 4; ++k) if (e[k] > e[i0]) i0 = k;
  int i1 = -1;
  for (int k = 0; k < 4; ++k) { if (k == i0) continue; if (i1 < 0 || e[k] > e[i1]) i1 = k; }
  float den = e[i0] + e[i1];
  for (int k = 0; k < 4; ++k)
    comb[t * 4 + k] = (k == i0) ? e[i0] / den : ((k == i1) ? e[i1] / den : 0.f);
}

__global__ void moe_combine(const __bf16* __restrict__ y0, const __bf16* __restrict__ y1,
                            const __bf16* __restrict__ y2, const __bf16* __restrict__ y3,
                            const float* __restrict__ comb, __bf16* __restrict__ outp,
                            int N, int C) {
  int t = blockIdx.x * blockDim.x + threadIdx.x;
  if (t >= N * C) return;
  int n = t / C;
  float v = comb[n * 4 + 0] * (float)y0[t] + comb[n * 4 + 1] * (float)y1[t] +
            comb[n * 4 + 2] * (float)y2[t] + comb[n * 4 + 3] * (float)y3[t];
  outp[t] = (__bf16)v;
}

__global__ void pshuffle2(const __bf16* __restrict__ in, __bf16* __restrict__ outp,
                          int H, int W, int C) {
  // in [H][W][4C] -> out [2H][2W][C]; out ch c at (2y+r1,2x+r2) = in ch c*4+r1*2+r2
  int t = blockIdx.x * blockDim.x + threadIdx.x;
  int total = 4 * H * W * C;
  if (t >= total) return;
  int c = t % C;
  int rest = t / C;
  int W2 = 2 * W;
  int ox = rest % W2;
  int oy = rest / W2;
  int iy = oy >> 1, r1 = oy & 1, ix = ox >> 1, r2 = ox & 1;
  outp[t] = in[((size_t)iy * W + ix) * (4 * C) + c * 4 + r1 * 2 + r2];
}

__global__ void add_resize4(const float* __restrict__ lr, float* __restrict__ io, int h, int w) {
  int t = blockIdx.x * blockDim.x + threadIdx.x;
  int W4 = w * 4, H4 = h * 4;
  int total = 3 * H4 * W4;
  if (t >= total) return;
  int c = t / (H4 * W4);
  int p = t - c * (H4 * W4);
  int oy = p / W4, ox = p - oy * W4;
  float sy = ((float)oy + 0.5f) * 0.25f - 0.5f;
  float sx = ((float)ox + 0.5f) * 0.25f - 0.5f;
  float y0f = floorf(sy), x0f = floorf(sx);
  float wy = sy - y0f, wx = sx - x0f;
  int y0 = (int)y0f, x0 = (int)x0f;
  int y0c = y0 < 0 ? 0 : (y0 > h - 1 ? h - 1 : y0);
  int y1c = y0 + 1 < 0 ? 0 : (y0 + 1 > h - 1 ? h - 1 : y0 + 1);
  int x0c = x0 < 0 ? 0 : (x0 > w - 1 ? w - 1 : x0);
  int x1c = x0 + 1 < 0 ? 0 : (x0 + 1 > w - 1 ? w - 1 : x0 + 1);
  const float* ch = lr + (size_t)c * h * w;
  float v = ch[y0c * w + x0c] * (1.f - wy) * (1.f - wx) + ch[y0c * w + x1c] * (1.f - wy) * wx +
            ch[y1c * w + x0c] * wy * (1.f - wx) + ch[y1c * w + x1c] * wy * wx;
  io[t] += v;
}

// ------------------------------ host side ----------------------------------
static void conv3x3(hipStream_t s, const void* in, const void* wp, const float* bias,
                    void* outb, float* outf, const void* res,
                    int H, int W, int Cinp, int Cout, int Coutp, int ldo, int act) {
  dim3 grid(H * (W / 64));
  size_t shm = (size_t)3 * 66 * Cinp * sizeof(__bf16);
#define C3(ACT, RES, OB, OF)                                                     \
  conv3x3_wmma<ACT, RES, OB, OF><<<grid, dim3(256), shm, s>>>(                   \
      (const __bf16*)in, (const __bf16*)wp, bias, (__bf16*)outb, outf,           \
      (const __bf16*)res, H, W, Cinp, Cout, Coutp, ldo)
  if (outf)          C3(0, false, false, true);
  else if (res)      C3(0, true, true, false);
  else if (act == 1) C3(1, false, true, false);
  else if (act == 2) C3(2, false, true, false);
  else               C3(0, false, true, false);
#undef C3
}

static void gemmL(hipStream_t s, const void* X, const void* Wt, const float* bias,
                  void* outb, int N, int Kp, int M, int Mp, int ldo, int co_off, int act) {
  int waves = (Mp / 16) * (N / 32);
  dim3 grid((waves + 7) / 8);
  if (act == 1)
    gemm_wmma<1><<<grid, dim3(256), 0, s>>>((const __bf16*)X, (const __bf16*)Wt, bias,
                                            (__bf16*)outb, N, Kp, M, Mp, ldo, co_off);
  else if (act == 3)
    gemm_wmma<3><<<grid, dim3(256), 0, s>>>((const __bf16*)X, (const __bf16*)Wt, bias,
                                            (__bf16*)outb, N, Kp, M, Mp, ldo, co_off);
  else
    gemm_wmma<0><<<grid, dim3(256), 0, s>>>((const __bf16*)X, (const __bf16*)Wt, bias,
                                            (__bf16*)outb, N, Kp, M, Mp, ldo, co_off);
}

extern "C" void kernel_launch(void* const* d_in, const int* in_sizes, int n_in,
                              void* d_out, int out_size, void* d_ws, size_t ws_size,
                              hipStream_t stream) {
  (void)in_sizes; (void)n_in; (void)out_size; (void)ws_size;
  const float* lrs    = (const float*)d_in[0];
  const float* cf_w   = (const float*)d_in[1];
  const float* cf_b   = (const float*)d_in[2];
  const float* fw1    = (const float*)d_in[3];
  const float* fb1    = (const float*)d_in[4];
  const float* fw2    = (const float*)d_in[5];
  const float* fb2    = (const float*)d_in[6];
  const float* bwd_wi = (const float*)d_in[7];
  const float* bwd_bi = (const float*)d_in[8];
  const float* bwd_w1 = (const float*)d_in[9];
  const float* bwd_b1 = (const float*)d_in[10];
  const float* bwd_w2 = (const float*)d_in[11];
  const float* bwd_b2 = (const float*)d_in[12];
  const float* fwd_wi = (const float*)d_in[13];
  const float* fwd_bi = (const float*)d_in[14];
  const float* fwd_w1 = (const float*)d_in[15];
  const float* fwd_b1 = (const float*)d_in[16];
  const float* fwd_w2 = (const float*)d_in[17];
  const float* fwd_b2 = (const float*)d_in[18];
  const float* pc_w   = (const float*)d_in[19];
  const float* pc_b   = (const float*)d_in[20];
  const float* up1_w  = (const float*)d_in[21];
  const float* up1_b  = (const float*)d_in[22];
  const float* up2_w  = (const float*)d_in[23];
  const float* up2_b  = (const float*)d_in[24];
  const float* cl1_w  = (const float*)d_in[25];
  const float* cl1_b  = (const float*)d_in[26];
  const float* cl2_w  = (const float*)d_in[27];
  const float* cl2_b  = (const float*)d_in[28];
  const float* co1_w  = (const float*)d_in[29];
  const float* co1_b  = (const float*)d_in[30];
  const float* co2_w  = (const float*)d_in[31];
  const float* co2_b  = (const float*)d_in[32];
  const float* co3_w  = (const float*)d_in[33];
  const float* co3_b  = (const float*)d_in[34];
  const float* co4_w  = (const float*)d_in[35];
  const float* co4_b  = (const float*)d_in[36];
  const float* dcn_w  = (const float*)d_in[37];
  const float* dcn_b  = (const float*)d_in[38];
  const float* gate_w = (const float*)d_in[39];
  const float* ew1    = (const float*)d_in[40];
  const float* eb1    = (const float*)d_in[41];
  const float* ew2    = (const float*)d_in[42];
  const float* eb2    = (const float*)d_in[43];
  float* dout = (float*)d_out;

  char* wsbase = (char*)d_ws;
  size_t off = 0;
  auto alloc = [&](size_t bytes) -> void* {
    void* p = wsbase + off;
    off = (off + bytes + 255) & ~(size_t)255;
    return p;
  };
  auto ab = [&](size_t elems) -> __bf16* { return (__bf16*)alloc(elems * 2); };
  auto af = [&](size_t elems) -> float*  { return (float*)alloc(elems * 4); };
  auto g1 = [](int n) { return dim3((unsigned)((n + 255) / 256)); };

  // activation / temp buffers
  __bf16* lrsb    = ab((size_t)6 * HWN * 32);
  __bf16* feats   = ab((size_t)6 * HWN * 64);
  __bf16* outsB   = ab((size_t)6 * HWN * 64);
  float*  flowsB  = af((size_t)5 * 2 * HWN);
  float*  flowsF  = af((size_t)5 * 2 * HWN);
  __bf16* flowin  = ab((size_t)HWN * 32);
  __bf16* fwtmp   = ab((size_t)HWN * 32);
  __bf16* warpout = ab((size_t)HWN * 64);
  __bf16* co1in   = ab((size_t)HWN * 160);
  __bf16* cobuf1  = ab((size_t)HWN * 64);
  __bf16* cobuf2  = ab((size_t)HWN * 64);
  float*  co4out  = af((size_t)108 * HWN);
  __bf16* samp    = ab((size_t)HWN * 576);
  __bf16* tokbuf  = ab((size_t)HWN * 96);
  __bf16* rbin    = ab((size_t)HWN * 96);
  float*  comb    = af((size_t)HWN * 4);
  __bf16* h1      = ab((size_t)HWN * 128);
  __bf16* yebuf   = ab((size_t)4 * HWN * 96);
  __bf16* rb_a    = ab((size_t)HWN * 64);
  __bf16* rb_b    = ab((size_t)HWN * 64);
  __bf16* featprop= ab((size_t)HWN * 64);
  __bf16* cat128  = ab((size_t)HWN * 128);
  __bf16* pcout   = ab((size_t)HWN * 64);
  __bf16* up1out  = ab((size_t)HWN * 256);
  __bf16* ps1     = ab((size_t)4 * HWN * 64);
  __bf16* up2out  = ab((size_t)4 * HWN * 256);
  __bf16* ps2     = ab((size_t)16 * HWN * 64);
  __bf16* cl1out  = ab((size_t)16 * HWN * 64);
  // packed weights (bf16)
  __bf16* cf_wp   = ab((size_t)64 * 288);
  __bf16* fw1p    = ab((size_t)32 * 288);
  __bf16* fw2p    = ab((size_t)16 * 288);
  __bf16* bwd_wip = ab((size_t)64 * 864);
  __bf16* fwd_wip = ab((size_t)64 * 864);
  __bf16* bwd_w1p = ab((size_t)8 * 64 * 576);
  __bf16* bwd_w2p = ab((size_t)8 * 64 * 576);
  __bf16* fwd_w1p = ab((size_t)8 * 64 * 576);
  __bf16* fwd_w2p = ab((size_t)8 * 64 * 576);
  __bf16* pc_wp   = ab((size_t)64 * 128);
  __bf16* up1_wp  = ab((size_t)256 * 576);
  __bf16* up2_wp  = ab((size_t)256 * 576);
  __bf16* cl1_wp  = ab((size_t)64 * 576);
  __bf16* cl2_wp  = ab((size_t)16 * 576);
  __bf16* co1_wp  = ab((size_t)64 * 1440);
  __bf16* co2_wp  = ab((size_t)64 * 576);
  __bf16* co3_wp  = ab((size_t)64 * 576);
  __bf16* co4_wp  = ab((size_t)112 * 576);
  __bf16* dcn_wp  = ab((size_t)64 * 576);
  __bf16* ew1p    = ab((size_t)4 * 128 * 96);
  __bf16* ew2p    = ab((size_t)4 * 80 * 128);

  // ---- pack all weights to bf16 ----
  pack_conv<<<g1(64 * 288), 256, 0, stream>>>(cf_w, cf_wp, 64, 3, 64, 32);
  pack_conv<<<g1(32 * 288), 256, 0, stream>>>(fw1, fw1p, 32, 6, 32, 32);
  pack_conv<<<g1(16 * 288), 256, 0, stream>>>(fw2, fw2p, 2, 32, 16, 32);
  pack_conv<<<g1(64 * 864), 256, 0, stream>>>(bwd_wi, bwd_wip, 64, 67, 64, 96);
  pack_conv<<<g1(64 * 864), 256, 0, stream>>>(fwd_wi, fwd_wip, 64, 67, 64, 96);
  for (int j = 0; j < 8; ++j) {
    pack_conv<<<g1(64 * 576), 256, 0, stream>>>(bwd_w1 + (size_t)j * 64 * 64 * 9, bwd_w1p + (size_t)j * 64 * 576, 64, 64, 64, 64);
    pack_conv<<<g1(64 * 576), 256, 0, stream>>>(bwd_w2 + (size_t)j * 64 * 64 * 9, bwd_w2p + (size_t)j * 64 * 576, 64, 64, 64, 64);
    pack_conv<<<g1(64 * 576), 256, 0, stream>>>(fwd_w1 + (size_t)j * 64 * 64 * 9, fwd_w1p + (size_t)j * 64 * 576, 64, 64, 64, 64);
    pack_conv<<<g1(64 * 576), 256, 0, stream>>>(fwd_w2 + (size_t)j * 64 * 64 * 9, fwd_w2p + (size_t)j * 64 * 576, 64, 64, 64, 64);
  }
  pack_mat<<<g1(64 * 128), 256, 0, stream>>>(pc_w, pc_wp, 64, 128, 64, 128, 128, 1);
  pack_conv<<<g1(256 * 576), 256, 0, stream>>>(up1_w, up1_wp, 256, 64, 256, 64);
  pack_conv<<<g1(256 * 576), 256, 0, stream>>>(up2_w, up2_wp, 256, 64, 256, 64);
  pack_conv<<<g1(64 * 576), 256, 0, stream>>>(cl1_w, cl1_wp, 64, 64, 64, 64);
  pack_conv<<<g1(16 * 576), 256, 0, stream>>>(cl2_w, cl2_wp, 3, 64, 16, 64);
  pack_conv<<<g1(64 * 1440), 256, 0, stream>>>(co1_w, co1_wp, 64, 130, 64, 160);
  pack_conv<<<g1(64 * 576), 256, 0, stream>>>(co2_w, co2_wp, 64, 64, 64, 64);
  pack_conv<<<g1(64 * 576), 256, 0, stream>>>(co3_w, co3_wp, 64, 64, 64, 64);
  pack_conv<<<g1(112 * 576), 256, 0, stream>>>(co4_w, co4_wp, 108, 64, 112, 64);
  pack_dcn<<<g1(64 * 576), 256, 0, stream>>>(dcn_w, dcn_wp);
  for (int e = 0; e < 4; ++e) {
    pack_mat<<<g1(128 * 96), 256, 0, stream>>>(ew1 + (size_t)e * 67 * 128, ew1p + (size_t)e * 128 * 96, 128, 67, 128, 96, 1, 128);
    pack_mat<<<g1(80 * 128), 256, 0, stream>>>(ew2 + (size_t)e * 128 * 67, ew2p + (size_t)e * 80 * 128, 67, 128, 80, 128, 1, 67);
  }

  // ---- zero padded concat buffers once per launch ----
  zero_bf16k<<<g1(6 * HWN * 32), 256, 0, stream>>>(lrsb, 6 * HWN * 32);
  zero_bf16k<<<g1(HWN * 32), 256, 0, stream>>>(flowin, HWN * 32);
  zero_bf16k<<<g1(HWN * 160), 256, 0, stream>>>(co1in, HWN * 160);
  zero_bf16k<<<g1(HWN * 96), 256, 0, stream>>>(tokbuf, HWN * 96);
  zero_bf16k<<<g1(HWN * 96), 256, 0, stream>>>(rbin, HWN * 96);
  zero_bf16k<<<g1(4 * HWN * 96), 256, 0, stream>>>(yebuf, 4 * HWN * 96);

  // ---- lrs to bf16 HWC, feature extraction, flows ----
  for (int f = 0; f < 6; ++f)
    f2h_hwc<<<g1(3 * HWN), 256, 0, stream>>>(lrs + (size_t)f * 3 * HWN, lrsb + (size_t)f * HWN * 32, 3, HWN, 32, 0);
  for (int f = 0; f < 6; ++f)
    conv3x3(stream, lrsb + (size_t)f * HWN * 32, cf_wp, cf_b, feats + (size_t)f * HWN * 64,
            nullptr, nullptr, 64, 64, 32, 64, 64, 64, 0);
  for (int i = 0; i < 5; ++i) {
    f2h_hwc<<<g1(3 * HWN), 256, 0, stream>>>(lrs + (size_t)i * 3 * HWN, flowin, 3, HWN, 32, 0);
    f2h_hwc<<<g1(3 * HWN), 256, 0, stream>>>(lrs + (size_t)(i + 1) * 3 * HWN, flowin, 3, HWN, 32, 3);
    conv3x3(stream, flowin, fw1p, fb1, fwtmp, nullptr, nullptr, 64, 64, 32, 32, 32, 32, 2);
    conv3x3(stream, fwtmp, fw2p, fb2, nullptr, flowsB + (size_t)i * 2 * HWN, nullptr, 64, 64, 32, 2, 16, 0, 0);
    f2h_hwc<<<g1(3 * HWN), 256, 0, stream>>>(lrs + (size_t)(i + 1) * 3 * HWN, flowin, 3, HWN, 32, 0);
    f2h_hwc<<<g1(3 * HWN), 256, 0, stream>>>(lrs + (size_t)i * 3 * HWN, flowin, 3, HWN, 32, 3);
    conv3x3(stream, flowin, fw1p, fb1, fwtmp, nullptr, nullptr, 64, 64, 32, 32, 32, 32, 2);
    conv3x3(stream, fwtmp, fw2p, fb2, nullptr, flowsF + (size_t)i * 2 * HWN, nullptr, 64, 64, 32, 2, 16, 0, 0);
  }

  auto pa_moe = [&](const __bf16* x_samp, const __bf16* x_cur, const float* fl, const float* lrs_i) {
    hwc_copy<<<g1(64 * HWN), 256, 0, stream>>>(warpout, co1in, 64, HWN, 64, 0, 160, 0);
    hwc_copy<<<g1(64 * HWN), 256, 0, stream>>>(x_cur, co1in, 64, HWN, 64, 0, 160, 64);
    f2h_hwc<<<g1(2 * HWN), 256, 0, stream>>>(fl, co1in, 2, HWN, 160, 128);
    conv3x3(stream, co1in, co1_wp, co1_b, cobuf1, nullptr, nullptr, 64, 64, 160, 64, 64, 64, 1);
    conv3x3(stream, cobuf1, co2_wp, co2_b, cobuf2, nullptr, nullptr, 64, 64, 64, 64, 64, 64, 1);
    conv3x3(stream, cobuf2, co3_wp, co3_b, cobuf1, nullptr, nullptr, 64, 64, 64, 64, 64, 64, 1);
    conv3x3(stream, cobuf1, co4_wp, co4_b, nullptr, co4out, nullptr, 64, 64, 64, 108, 112, 0, 0);
    dcn_sample<<<g1(HWN * 36), 256, 0, stream>>>(x_samp, co4out, fl, samp, 64, 64);
    gemmL(stream, samp, dcn_wp, dcn_b, tokbuf, HWN, 576, 64, 64, 96, 3, 0);
    f2h_hwc<<<g1(3 * HWN), 256, 0, stream>>>(lrs_i, tokbuf, 3, HWN, 96, 0);
    moe_gate<<<g1(HWN), 256, 0, stream>>>(tokbuf, gate_w, comb, HWN, 67, 96);
    for (int e = 0; e < 4; ++e) {
      gemmL(stream, tokbuf, ew1p + (size_t)e * 128 * 96, eb1 + e * 128, h1, HWN, 96, 128, 128, 128, 0, 3);
      gemmL(stream, h1, ew2p + (size_t)e * 80 * 128, eb2 + e * 67, yebuf + (size_t)e * HWN * 96, HWN, 128, 67, 80, 96, 0, 0);
    }
    moe_combine<<<g1(HWN * 96), 256, 0, stream>>>(yebuf, yebuf + (size_t)HWN * 96, yebuf + (size_t)2 * HWN * 96,
                                                  yebuf + (size_t)3 * HWN * 96, comb, rbin, HWN, 96);
  };

  auto resblocks = [&](const __bf16* inb, const __bf16* wip, const float* bi,
                       const __bf16* w1p, const float* b1, const __bf16* w2p, const float* b2) {
    conv3x3(stream, inb, wip, bi, rb_a, nullptr, nullptr, 64, 64, 96, 64, 64, 64, 1);
    for (int j = 0; j < 8; ++j) {
      conv3x3(stream, rb_a, w1p + (size_t)j * 64 * 576, b1 + j * 64, rb_b, nullptr, nullptr, 64, 64, 64, 64, 64, 64, 2);
      conv3x3(stream, rb_b, w2p + (size_t)j * 64 * 576, b2 + j * 64, rb_a, nullptr, rb_a, 64, 64, 64, 64, 64, 64, 0);
    }
  };

  // ---- backward pass ----
  for (int i = 5; i >= 0; --i) {
    if (i == 5) {
      f2h_hwc<<<g1(3 * HWN), 256, 0, stream>>>(lrs + (size_t)i * 3 * HWN, rbin, 3, HWN, 96, 0);
      hwc_copy<<<g1(64 * HWN), 256, 0, stream>>>(feats + (size_t)i * HWN * 64, rbin, 64, HWN, 64, 0, 96, 3);
    } else {
      const float* fl = flowsB + (size_t)i * 2 * HWN;
      warp_bilinear<<<g1(HWN * 8), 256, 0, stream>>>(feats + (size_t)i * HWN * 64, fl, warpout, 64, 64, 64, 64, 64);
      pa_moe(feats + (size_t)i * HWN * 64, feats + (size_t)(i + 1) * HWN * 64, fl, lrs + (size_t)i * 3 * HWN);
    }
    resblocks(rbin, bwd_wip, bwd_bi, bwd_w1p, bwd_b1, bwd_w2p, bwd_b2);
    hwc_copy<<<g1(64 * HWN), 256, 0, stream>>>(rb_a, outsB + (size_t)i * HWN * 64, 64, HWN, 64, 0, 64, 0);
  }

  // ---- forward pass + upsampling ----
  hwc_copy<<<g1(64 * HWN), 256, 0, stream>>>(feats, featprop, 64, HWN, 64, 0, 64, 0);
  for (int i = 0; i < 6; ++i) {
    if (i == 0) {
      f2h_hwc<<<g1(3 * HWN), 256, 0, stream>>>(lrs, rbin, 3, HWN, 96, 0);
      hwc_copy<<<g1(64 * HWN), 256, 0, stream>>>(featprop, rbin, 64, HWN, 64, 0, 96, 3);
    } else {
      const float* fl = flowsF + (size_t)(i - 1) * 2 * HWN;
      warp_bilinear<<<g1(HWN * 8), 256, 0, stream>>>(featprop, fl, warpout, 64, 64, 64, 64, 64);
      pa_moe(feats + (size_t)(i - 1) * HWN * 64, feats + (size_t)i * HWN * 64, fl, lrs + (size_t)i * 3 * HWN);
    }
    resblocks(rbin, fwd_wip, fwd_bi, fwd_w1p, fwd_b1, fwd_w2p, fwd_b2);
    hwc_copy<<<g1(64 * HWN), 256, 0, stream>>>(rb_a, featprop, 64, HWN, 64, 0, 64, 0);

    hwc_copy<<<g1(64 * HWN), 256, 0, stream>>>(outsB + (size_t)i * HWN * 64, cat128, 64, HWN, 64, 0, 128, 0);
    hwc_copy<<<g1(64 * HWN), 256, 0, stream>>>(featprop, cat128, 64, HWN, 64, 0, 128, 64);
    gemmL(stream, cat128, pc_wp, pc_b, pcout, HWN, 128, 64, 64, 64, 0, 1);
    conv3x3(stream, pcout, up1_wp, up1_b, up1out, nullptr, nullptr, 64, 64, 64, 256, 256, 256, 0);
    pshuffle2<<<g1(4 * HWN * 64), 256, 0, stream>>>(up1out, ps1, 64, 64, 64);
    conv3x3(stream, ps1, up2_wp, up2_b, up2out, nullptr, nullptr, 128, 128, 64, 256, 256, 256, 0);
    pshuffle2<<<g1(16 * HWN * 64), 256, 0, stream>>>(up2out, ps2, 128, 128, 64);
    conv3x3(stream, ps2, cl1_wp, cl1_b, cl1out, nullptr, nullptr, 256, 256, 64, 64, 64, 64, 1);
    float* outframe = dout + (size_t)i * 3 * 65536;
    conv3x3(stream, cl1out, cl2_wp, cl2_b, nullptr, outframe, nullptr, 256, 256, 64, 3, 16, 0, 0);
    add_resize4<<<g1(3 * 65536), 256, 0, stream>>>(lrs + (size_t)i * 3 * HWN, outframe, 64, 64);
  }
}
